// Qwen3_5MoeVisionModel_44547400794844
// MI455X (gfx1250) — matrix operand order repaired and checked by reading, not compiled, not run on hardware
//
#include <hip/hip_runtime.h>

// ---------------------------------------------------------------------------
// Qwen3.5 vision tower forward for MI455X (gfx1250), bf16 WMMA everywhere.
// All WMMA fragments are fed with aligned 16B (float4 / b128) loads.
// ---------------------------------------------------------------------------

typedef __bf16 bf16;
typedef __bf16 v16bf __attribute__((ext_vector_type(16)));
typedef float  v8f   __attribute__((ext_vector_type(8)));

#define T_TOK   1280
#define HID     1152
#define NHEAD   16
#define HDIM    72
#define HDP     96      // padded head dim for Q/K (K-dim of QK^T, mult of 32)
#define HDV     80      // padded head dim for V / O columns (mult of 16)
#define INTER_R 4304
#define INTER_P 4352    // padded (mult of 64)
#define ATTN_SCALE 0.11785113019775793f  // 72^-0.5

struct Two4 { float4 lo, hi; };

// Build a 16-element bf16 fragment from two aligned 16B chunks.
// A-fragment (ISA 7.12.2, 16-bit A 16x32): elems 0..7 <- K[half*8 .. +8),
//   elems 8..15 <- K[16+half*8 .. +8)  -> p0 = base+half*8, p1 = base+16+half*8
// B-fragment (same order as A):            elems 0..7 <- K[half*8 .. +8),
//   elems 8..15 <- K[16+half*8 .. +8)  -> p0 = base+half*8, p1 = base+16+half*8
__device__ __forceinline__ v16bf frag_from(const bf16* p0, const bf16* p1) {
  Two4 t;
  t.lo = *reinterpret_cast<const float4*>(p0);
  t.hi = *reinterpret_cast<const float4*>(p1);
  return __builtin_bit_cast(v16bf, t);
}
__device__ __forceinline__ v16bf frag_a(const bf16* base, int half) {
  return frag_from(base + (half << 3), base + 16 + (half << 3));
}
__device__ __forceinline__ v16bf frag_b(const bf16* base, int half) {
  return frag_from(base + (half << 3), base + 16 + (half << 3));
}

__device__ __forceinline__ float gelu_tanh_f(float x) {
  float x3 = x * x * x;
  return 0.5f * x * (1.0f + tanhf(0.7978845608028654f * (x + 0.044715f * x3)));
}
__device__ __forceinline__ float gelu_erf_f(float x) {
  return 0.5f * x * (1.0f + erff(x * 0.7071067811865476f));
}

// token index -> (row, col) position in the 2x2-merged grid ordering
__device__ __forceinline__ void token_pos(int t, int& rowp, int& colp) {
  int n, w;
  if (t < 1024) { n = t; w = 32; }
  else          { n = t - 1024; w = 16; }
  int wi = n & 1, hi = (n >> 1) & 1, rest = n >> 2;
  int wb = rest % (w >> 1), hb = rest / (w >> 1);
  rowp = hb * 2 + hi;
  colp = wb * 2 + wi;
}

// ---------------------------------------------------------------------------
// f32 -> bf16 convert with zero padding: dst[prows x pcols] from src[rows x cols]
// ---------------------------------------------------------------------------
__global__ void cvt_pad_kernel(const float* __restrict__ src, bf16* __restrict__ dst,
                               int rows, int cols, int pcols, size_t total) {
  size_t idx = (size_t)blockIdx.x * blockDim.x + threadIdx.x;
  size_t stride = (size_t)gridDim.x * blockDim.x;
  for (; idx < total; idx += stride) {
    int r = (int)(idx / pcols);
    int c = (int)(idx - (size_t)r * pcols);
    float v = (r < rows && c < cols) ? src[(size_t)r * cols + c] : 0.0f;
    dst[idx] = (bf16)v;
  }
}

// ---------------------------------------------------------------------------
// Generic bf16 WMMA GEMM:  out[M x N] = A[M x Kpad] * W[N x Kpad]^T + bias
// Block = 128 threads = 4 waves, tile 64x64, K step 32.
// EPI: 0 = f32 + bias; 1 = f32 + bias + residual; 2 = bf16 gelu_tanh + zero-pad;
//      3 = bf16 gelu_erf
// ---------------------------------------------------------------------------
template <int EPI>
__global__ __launch_bounds__(128) void gemm_wmma(
    const bf16* __restrict__ A, const bf16* __restrict__ B,
    const float* __restrict__ bias, void* outp, const float* resid,
    int ldo, int Nreal, int Kpad) {
  __shared__ bf16 As[64][40];   // 80B rows -> 16B aligned float4 access
  __shared__ bf16 Bs[64][40];
  const int tid  = threadIdx.x;
  const int lane = tid & 31;
  const int wave = tid >> 5;
  const int wm = wave >> 1, wn = wave & 1;
  const int half = lane >> 4, l15 = lane & 15;
  const int m0 = blockIdx.y * 64;
  const int n0 = blockIdx.x * 64;

  v8f acc[2][2];
#pragma unroll
  for (int a = 0; a < 2; ++a)
#pragma unroll
    for (int b = 0; b < 2; ++b)
#pragma unroll
      for (int r = 0; r < 8; ++r) acc[a][b][r] = 0.0f;

  for (int k0 = 0; k0 < Kpad; k0 += 32) {
    // cooperative load of 64x32 A and B tiles, one aligned float4 per chunk
#pragma unroll
    for (int e = tid; e < 256; e += 128) {
      int row = e >> 2;
      int col = (e & 3) << 3;   // 0,8,16,24 -> 16B aligned in LDS (stride 80B)
      *reinterpret_cast<float4*>(&As[row][col]) =
          *reinterpret_cast<const float4*>(A + (size_t)(m0 + row) * Kpad + k0 + col);
      *reinterpret_cast<float4*>(&Bs[row][col]) =
          *reinterpret_cast<const float4*>(B + (size_t)(n0 + row) * Kpad + k0 + col);
    }
    if (k0 + 32 < Kpad) {  // -> global_prefetch_b8 for next K tile
      __builtin_prefetch(A + (size_t)(m0 + (tid >> 1)) * Kpad + k0 + 32, 0, 1);
      __builtin_prefetch(B + (size_t)(n0 + (tid >> 1)) * Kpad + k0 + 32, 0, 1);
    }
    __syncthreads();

    v16bf af[2], bfr[2];
#pragma unroll
    for (int s = 0; s < 2; ++s) {
      af[s]  = frag_a(&As[wm * 32 + s * 16 + l15][0], half);
      bfr[s] = frag_b(&Bs[wn * 32 + s * 16 + l15][0], half);
    }
#pragma unroll
    for (int a = 0; a < 2; ++a)
#pragma unroll
      for (int b = 0; b < 2; ++b)
        acc[a][b] = __builtin_amdgcn_wmma_f32_16x16x32_bf16(
            false, af[a], false, bfr[b], (short)0, acc[a][b], false, false);
    __syncthreads();
  }

#pragma unroll
  for (int a = 0; a < 2; ++a)
#pragma unroll
    for (int b = 0; b < 2; ++b)
#pragma unroll
      for (int r = 0; r < 8; ++r) {
        int m = m0 + wm * 32 + a * 16 + r + (half << 3);
        int n = n0 + wn * 32 + b * 16 + l15;
        float v = acc[a][b][r];
        if constexpr (EPI == 0 || EPI == 1) {
          v += bias[n];
          if constexpr (EPI == 1) v += resid[(size_t)m * ldo + n];
          reinterpret_cast<float*>(outp)[(size_t)m * ldo + n] = v;
        } else {
          float o = 0.0f;
          if (n < Nreal) {
            v += bias[n];
            o = (EPI == 2) ? gelu_tanh_f(v) : gelu_erf_f(v);
          }
          reinterpret_cast<bf16*>(outp)[(size_t)m * ldo + n] = (bf16)o;
        }
      }
}

// ---------------------------------------------------------------------------
// LayerNorm over HID, output bf16 (always feeds a GEMM)
// ---------------------------------------------------------------------------
__global__ __launch_bounds__(128) void ln_bf16(const float* __restrict__ x,
                                               const float* __restrict__ g,
                                               const float* __restrict__ b,
                                               bf16* __restrict__ out, int ldo) {
  const int t = blockIdx.x;
  const int tid = threadIdx.x;
  __shared__ float s1[128], s2[128];
  const float* xr = x + (size_t)t * HID;
  float a = 0.0f, q = 0.0f;
  for (int c = tid; c < HID; c += 128) { float v = xr[c]; a += v; q += v * v; }
  s1[tid] = a; s2[tid] = q;
  __syncthreads();
  for (int s = 64; s > 0; s >>= 1) {
    if (tid < s) { s1[tid] += s1[tid + s]; s2[tid] += s2[tid + s]; }
    __syncthreads();
  }
  float mean = s1[0] * (1.0f / HID);
  float var  = s2[0] * (1.0f / HID) - mean * mean;
  float rstd = rsqrtf(var + 1e-6f);
  for (int c = tid; c < HID; c += 128)
    out[(size_t)t * ldo + c] = (bf16)((xr[c] - mean) * rstd * g[c] + b[c]);
}

// ---------------------------------------------------------------------------
// Bilinear position-embedding interpolation, added in-place to h
// ---------------------------------------------------------------------------
__global__ void pos_add_kernel(const float* __restrict__ pe, float* __restrict__ hb) {
  int idx = blockIdx.x * blockDim.x + threadIdx.x;
  if (idx >= T_TOK * HID) return;
  int t = idx / HID;
  int c = idx - t * HID;
  int hp, wp;
  token_pos(t, hp, wp);
  int hh = (t < 1024) ? 32 : 16;
  double fh = 47.0 * hp / (hh - 1);
  double fw = 47.0 * wp / (hh - 1);
  int h0 = (int)floor(fh); int h1 = min(h0 + 1, 47);
  int w0 = (int)floor(fw); int w1 = min(w0 + 1, 47);
  float dh = (float)(fh - h0), dw = (float)(fw - w0);
  const float* base = pe + (size_t)c;
  float p00 = base[((size_t)h0 * 48 + w0) * HID];
  float p01 = base[((size_t)h0 * 48 + w1) * HID];
  float p10 = base[((size_t)h1 * 48 + w0) * HID];
  float p11 = base[((size_t)h1 * 48 + w1) * HID];
  hb[idx] += p00 * (1 - dh) * (1 - dw) + p01 * (1 - dh) * dw +
             p10 * dh * (1 - dw) + p11 * dh * dw;
}

// ---------------------------------------------------------------------------
// QKV split + RoPE -> padded bf16 Q[NH][T][96], K[NH][T][96], Vt[NH][80][T]
// ---------------------------------------------------------------------------
__global__ void rope_split(const float* __restrict__ qkv, bf16* __restrict__ qb,
                           bf16* __restrict__ kb, bf16* __restrict__ vt) {
  const int t = blockIdx.x;
  const int head = blockIdx.y;
  const int d = threadIdx.x;
  if (d >= HDIM) return;
  int rowp, colp;
  token_pos(t, rowp, colp);
  int d36 = d % 36;
  int j; float pos;
  if (d36 < 18) { j = d36;      pos = (float)rowp; }
  else          { j = d36 - 18; pos = (float)colp; }
  // inv = 10000^(-2j/36)
  float f = pos * __expf(-9.210340371976184f * (float)j / 18.0f);
  float cs = __cosf(f), sn = __sinf(f);
  const float* base = qkv + (size_t)t * (3 * HID);
  int col  = head * HDIM + d;
  int pcol = head * HDIM + (d < 36 ? d + 36 : d - 36);
  float sgn = (d < 36) ? -1.0f : 1.0f;
  float q0 = base[col],  k0 = base[HID + col], v0 = base[2 * HID + col];
  float qp = base[pcol], kp = base[HID + pcol];
  float qr = q0 * cs + sgn * qp * sn;
  float kr = k0 * cs + sgn * kp * sn;
  qb[((size_t)head * T_TOK + t) * HDP + d] = (bf16)qr;
  kb[((size_t)head * T_TOK + t) * HDP + d] = (bf16)kr;
  vt[((size_t)head * HDV + d) * T_TOK + t] = (bf16)v0;
}

// ---------------------------------------------------------------------------
// Flash-style windowed attention: one wave per (16-row q tile, head).
// QK^T and P*V via v_wmma_f32_16x16x32_bf16; online softmax with wave32
// half-group shuffles. All fragment feeds are aligned b128 loads.
// ---------------------------------------------------------------------------
__global__ __launch_bounds__(32) void attn_wmma(const bf16* __restrict__ qb,
                                                const bf16* __restrict__ kb,
                                                const bf16* __restrict__ vtb,
                                                bf16* __restrict__ outb) {
  const int qtile = blockIdx.x;
  const int head  = blockIdx.y;
  const int lane  = threadIdx.x;
  const int half  = lane >> 4, l15 = lane & 15;
  const int off   = (qtile < 64) ? 0 : 1024;
  const int L     = (qtile < 64) ? 1024 : 256;
  const int qbase = qtile * 16;

  __shared__ bf16 Ps[16][40];   // 80B rows -> aligned b128 A-fragment reads

  v16bf qa[3];
  {
    const bf16* qrow = qb + ((size_t)head * T_TOK + (qbase + l15)) * HDP;
#pragma unroll
    for (int ks = 0; ks < 3; ++ks) qa[ks] = frag_a(qrow + ks * 32, half);
  }

  v8f oc[5];
#pragma unroll
  for (int t = 0; t < 5; ++t)
#pragma unroll
    for (int r = 0; r < 8; ++r) oc[t][r] = 0.0f;
  float rm[8], rs[8];
#pragma unroll
  for (int r = 0; r < 8; ++r) { rm[r] = -1e30f; rs[r] = 0.0f; }

  for (int kv0 = 0; kv0 < L; kv0 += 32) {
    v8f c0, c1;
#pragma unroll
    for (int r = 0; r < 8; ++r) { c0[r] = 0.0f; c1[r] = 0.0f; }
#pragma unroll
    for (int ks = 0; ks < 3; ++ks) {
      const bf16* kr0 = kb + ((size_t)head * T_TOK + (off + kv0 + l15)) * HDP + ks * 32;
      v16bf kf0 = frag_b(kr0, half);
      v16bf kf1 = frag_b(kr0 + 16 * HDP, half);
      c0 = __builtin_amdgcn_wmma_f32_16x16x32_bf16(false, qa[ks], false, kf0, (short)0, c0, false, false);
      c1 = __builtin_amdgcn_wmma_f32_16x16x32_bf16(false, qa[ks], false, kf1, (short)0, c1, false, false);
    }
    float fac[8];
#pragma unroll
    for (int r = 0; r < 8; ++r) {
      float s0 = c0[r] * ATTN_SCALE;
      float s1 = c1[r] * ATTN_SCALE;
      float mx = fmaxf(s0, s1);
#pragma unroll
      for (int dl = 1; dl < 16; dl <<= 1) mx = fmaxf(mx, __shfl_xor(mx, dl, 32));
      float nm = fmaxf(rm[r], mx);
      float f  = __expf(rm[r] - nm);
      float p0 = __expf(s0 - nm);
      float p1 = __expf(s1 - nm);
      float ss = p0 + p1;
#pragma unroll
      for (int dl = 1; dl < 16; dl <<= 1) ss += __shfl_xor(ss, dl, 32);
      rs[r] = rs[r] * f + ss;
      rm[r] = nm;
      fac[r] = f;
      Ps[(half << 3) + r][l15]      = (bf16)p0;
      Ps[(half << 3) + r][16 + l15] = (bf16)p1;
    }
#pragma unroll
    for (int t = 0; t < 5; ++t)
#pragma unroll
      for (int r = 0; r < 8; ++r) oc[t][r] *= fac[r];
    __syncthreads();
    v16bf pa = frag_a(&Ps[l15][0], half);
#pragma unroll
    for (int t = 0; t < 5; ++t) {
      const bf16* vr = vtb + ((size_t)head * HDV + t * 16 + l15) * T_TOK + off + kv0;
      v16bf vf = frag_b(vr, half);
      oc[t] = __builtin_amdgcn_wmma_f32_16x16x32_bf16(false, pa, false, vf, (short)0, oc[t], false, false);
    }
    __syncthreads();
  }
#pragma unroll
  for (int r = 0; r < 8; ++r) {
    float inv = 1.0f / rs[r];
    int m = qbase + (half << 3) + r;
#pragma unroll
    for (int t = 0; t < 5; ++t) {
      int d = t * 16 + l15;
      if (d < HDIM)
        outb[(size_t)m * HID + head * HDIM + d] = (bf16)(oc[t][r] * inv);
    }
  }
}

// ---------------------------------------------------------------------------
// Host driver
// ---------------------------------------------------------------------------
extern "C" void kernel_launch(void* const* d_in, const int* in_sizes, int n_in,
                              void* d_out, int out_size, void* d_ws, size_t ws_size,
                              hipStream_t stream) {
  (void)in_sizes; (void)n_in; (void)out_size; (void)ws_size;
  // setup_inputs() recursive dict insertion order
  const float* pixel   = (const float*)d_in[0];
  const float* ln1_g   = (const float*)d_in[1];
  const float* ln1_b   = (const float*)d_in[2];
  const float* qkv_w   = (const float*)d_in[3];
  const float* qkv_b   = (const float*)d_in[4];
  const float* proj_w  = (const float*)d_in[5];
  const float* proj_b  = (const float*)d_in[6];
  const float* ln2_g   = (const float*)d_in[7];
  const float* ln2_b   = (const float*)d_in[8];
  const float* fc1_w   = (const float*)d_in[9];
  const float* fc1_b   = (const float*)d_in[10];
  const float* fc2_w   = (const float*)d_in[11];
  const float* fc2_b   = (const float*)d_in[12];
  const float* patch_w = (const float*)d_in[13];
  const float* patch_b = (const float*)d_in[14];
  const float* pos_t   = (const float*)d_in[15];
  const float* mln_g   = (const float*)d_in[16];
  const float* mln_b   = (const float*)d_in[17];
  const float* mfc1_w  = (const float*)d_in[18];
  const float* mfc1_b  = (const float*)d_in[19];
  const float* mfc2_w  = (const float*)d_in[20];
  const float* mfc2_b  = (const float*)d_in[21];

  char* ws = (char*)d_ws;
  size_t off = 0;
  auto alloc = [&](size_t bytes) -> void* {
    void* p = ws + off;
    off = (off + bytes + 255) & ~(size_t)255;
    return p;
  };
  float* hbuf  = (float*)alloc((size_t)T_TOK * HID * 4);
  float* g1    = (float*)alloc((size_t)T_TOK * 3 * HID * 4);
  bf16*  a_bf  = (bf16*) alloc((size_t)T_TOK * INTER_P * 2);
  bf16*  a2_bf = (bf16*) alloc((size_t)T_TOK * INTER_P * 2);
  bf16*  w_bf  = (bf16*) alloc((size_t)4608 * 4608 * 2);
  bf16*  qbuf  = (bf16*) alloc((size_t)NHEAD * T_TOK * HDP * 2);
  bf16*  kbuf  = (bf16*) alloc((size_t)NHEAD * T_TOK * HDP * 2);
  bf16*  vtbuf = (bf16*) alloc((size_t)NHEAD * HDV * T_TOK * 2);

  hipMemsetAsync(qbuf,  0, (size_t)NHEAD * T_TOK * HDP * 2, stream);
  hipMemsetAsync(kbuf,  0, (size_t)NHEAD * T_TOK * HDP * 2, stream);
  hipMemsetAsync(vtbuf, 0, (size_t)NHEAD * HDV * T_TOK * 2, stream);

  auto cvt = [&](const float* src, bf16* dst, int rows, int cols, int prows, int pcols) {
    size_t total = (size_t)prows * pcols;
    int blocks = (int)((total + 255) / 256);
    if (blocks > 65535) blocks = 65535;
    cvt_pad_kernel<<<blocks, 256, 0, stream>>>(src, dst, rows, cols, pcols, total);
  };

  // ---- patch embed ----
  cvt(pixel, a_bf, T_TOK, 1536, T_TOK, 1536);
  cvt(patch_w, w_bf, HID, 1536, HID, 1536);
  gemm_wmma<0><<<dim3(HID / 64, T_TOK / 64), 128, 0, stream>>>(
      a_bf, w_bf, patch_b, hbuf, nullptr, HID, HID, 1536);
  pos_add_kernel<<<(T_TOK * HID + 255) / 256, 256, 0, stream>>>(pos_t, hbuf);

  // ---- 27 transformer blocks ----
  for (int l = 0; l < 27; ++l) {
    // ln1 -> qkv
    ln_bf16<<<T_TOK, 128, 0, stream>>>(hbuf, ln1_g + (size_t)l * HID,
                                       ln1_b + (size_t)l * HID, a_bf, HID);
    cvt(qkv_w + (size_t)l * 3 * HID * HID, w_bf, 3 * HID, HID, 3 * HID, HID);
    gemm_wmma<0><<<dim3(3 * HID / 64, T_TOK / 64), 128, 0, stream>>>(
        a_bf, w_bf, qkv_b + (size_t)l * 3 * HID, g1, nullptr, 3 * HID, 3 * HID, HID);
    // rope + attention
    rope_split<<<dim3(T_TOK, NHEAD), 96, 0, stream>>>(g1, qbuf, kbuf, vtbuf);
    attn_wmma<<<dim3(80, NHEAD), 32, 0, stream>>>(qbuf, kbuf, vtbuf, a_bf);
    // proj (+residual)
    cvt(proj_w + (size_t)l * HID * HID, w_bf, HID, HID, HID, HID);
    gemm_wmma<1><<<dim3(HID / 64, T_TOK / 64), 128, 0, stream>>>(
        a_bf, w_bf, proj_b + (size_t)l * HID, hbuf, hbuf, HID, HID, HID);
    // ln2 -> fc1 (gelu tanh, bf16 out, zero-padded to 4352)
    ln_bf16<<<T_TOK, 128, 0, stream>>>(hbuf, ln2_g + (size_t)l * HID,
                                       ln2_b + (size_t)l * HID, a_bf, HID);
    cvt(fc1_w + (size_t)l * INTER_R * HID, w_bf, INTER_R, HID, INTER_P, HID);
    gemm_wmma<2><<<dim3(INTER_P / 64, T_TOK / 64), 128, 0, stream>>>(
        a_bf, w_bf, fc1_b + (size_t)l * INTER_R, a2_bf, nullptr, INTER_P, INTER_R, HID);
    // fc2 (+residual)
    cvt(fc2_w + (size_t)l * HID * INTER_R, w_bf, HID, INTER_R, HID, INTER_P);
    gemm_wmma<1><<<dim3(HID / 64, T_TOK / 64), 128, 0, stream>>>(
        a2_bf, w_bf, fc2_b + (size_t)l * HID, hbuf, hbuf, HID, HID, INTER_P);
  }

  // ---- merge head ----
  // per-token LN, then view (1280 x 1152) as (320 x 4608) row-major
  ln_bf16<<<T_TOK, 128, 0, stream>>>(hbuf, mln_g, mln_b, a_bf, HID);
  cvt(mfc1_w, w_bf, 4608, 4608, 4608, 4608);
  gemm_wmma<3><<<dim3(4608 / 64, 320 / 64), 128, 0, stream>>>(
      a_bf, w_bf, mfc1_b, a2_bf, nullptr, 4608, 4608, 4608);
  cvt(mfc2_w, w_bf, 3584, 4608, 3584, 4608);
  gemm_wmma<0><<<dim3(3584 / 64, 320 / 64), 128, 0, stream>>>(
      a2_bf, w_bf, mfc2_b, (float*)d_out, nullptr, 3584, 3584, 4608);
}
